// ParallelBlock_4131758538938
// MI455X (gfx1250) — matrix operand order repaired and checked by reading, not compiled, not run on hardware
//
#include <hip/hip_runtime.h>
#include <math.h>
#include <stdint.h>

typedef __attribute__((ext_vector_type(16))) _Float16 v16h;
typedef __attribute__((ext_vector_type(8)))  float    v8f;
typedef unsigned int u32x4 __attribute__((ext_vector_type(4)));
typedef int          i32x4 __attribute__((ext_vector_type(4)));
typedef int          i32x8 __attribute__((ext_vector_type(8)));

#define DEV_INLINE __device__ __forceinline__

// ---------------------------------------------------------------- f32 -> f16
__global__ void k_f32_to_f16(const float* __restrict__ in, _Float16* __restrict__ out, size_t n) {
    size_t i = (size_t)blockIdx.x * blockDim.x + threadIdx.x;
    if (i < n) out[i] = (_Float16)in[i];
}

// ---------------------------------------------------------------- CPE: depthwise 3x3 + bias + residual (cls passthrough)
__global__ void k_cpe(const float* __restrict__ x, const float* __restrict__ w,
                      const float* __restrict__ bias, float* __restrict__ xo,
                      int N, int H, int W) {
    int c  = threadIdx.x;        // 0..255
    int bn = blockIdx.x;         // b*N + n
    int b  = bn / N, n = bn % N;
    float v = x[(size_t)bn * 256 + c];
    if (n > 0) {
        int p = n - 1, yy = p / W, xx = p % W;
        float acc = bias[c];
        #pragma unroll
        for (int ky = 0; ky < 3; ky++) {
            int iy = yy + ky - 1;
            if (iy < 0 || iy >= H) continue;
            #pragma unroll
            for (int kx = 0; kx < 3; kx++) {
                int ix = xx + kx - 1;
                if (ix < 0 || ix >= W) continue;
                acc += x[((size_t)b * N + 1 + (size_t)iy * W + ix) * 256 + c] * w[c * 9 + ky * 3 + kx];
            }
        }
        v += acc;
    }
    xo[(size_t)bn * 256 + c] = v;
}

// ---------------------------------------------------------------- LayerNorm over C=256, fused output to f16
__global__ void k_ln_f16(const float* __restrict__ x, const float* __restrict__ g,
                         const float* __restrict__ bb, _Float16* __restrict__ out, size_t nrows) {
    int lane = threadIdx.x & 31;
    size_t row = (size_t)blockIdx.x * 8 + (threadIdx.x >> 5);
    if (row >= nrows) return;
    const float* xr = x + row * 256;
    float vals[8], s = 0.f, s2 = 0.f;
    #pragma unroll
    for (int i = 0; i < 8; i++) { float v = xr[lane + 32 * i]; vals[i] = v; s += v; s2 += v * v; }
    #pragma unroll
    for (int m = 16; m > 0; m >>= 1) { s += __shfl_xor(s, m, 32); s2 += __shfl_xor(s2, m, 32); }
    float mu = s * (1.f / 256.f);
    float var = s2 * (1.f / 256.f) - mu * mu;
    float rs = rsqrtf(var + 1e-5f);
    #pragma unroll
    for (int i = 0; i < 8; i++) {
        int c = lane + 32 * i;
        out[row * 256 + c] = (_Float16)((vals[i] - mu) * rs * g[c] + bb[c]);
    }
}

// ---------------------------------------------------------------- TDM 2D tile load: global (row-major, f16) -> LDS
// Descriptor per CDNA5 ISA ch.8: group0 {count,lds_addr,global_addr,type=2},
// group1 {data_size=2B, pad 16DW->+4DW (80B row pitch), dims, strides}.
#define TDM_PITCH 40   // f16 per padded LDS row (64B data + 16B pad)

DEV_INLINE void tdm_load_2d(uint32_t lds_off, const _Float16* gptr,
                            uint32_t tensor_d0, uint32_t tensor_d1,
                            uint32_t tile_d0, uint32_t tile_d1, uint64_t stride_elems) {
    uint64_t ga = (uint64_t)(uintptr_t)gptr;
    u32x4 g0;
    g0[0] = 1u;                                                  // count=1 (valid user desc)
    g0[1] = lds_off;                                             // LDS byte address
    g0[2] = (uint32_t)(ga & 0xFFFFFFFFu);                        // global_addr[31:0]
    g0[3] = (uint32_t)((ga >> 32) & 0x1FFFFFFu) | (2u << 30);    // global_addr[56:32], type=2
    i32x8 g1;
    g1[0] = (int)((1u << 16) | (1u << 20) | (3u << 22) | (3u << 25));
        // data_size=1(2B)@16 | pad_enable@20 | pad_interval=3(16DW)@22 | pad_amount=3(4DW)@25
    g1[1] = (int)((tensor_d0 & 0xFFFFu) << 16);                  // tensor_dim0[15:0] @ bits63:48
    g1[2] = (int)(((tensor_d0 >> 16) & 0xFFFFu) | ((tensor_d1 & 0xFFFFu) << 16));
    g1[3] = (int)(((tensor_d1 >> 16) & 0xFFFFu) | ((tile_d0 & 0xFFFFu) << 16));
    g1[4] = (int)(tile_d1 & 0xFFFFu);                            // tile_dim1; tile_dim2=0
    g1[5] = (int)(stride_elems & 0xFFFFFFFFu);                   // tensor_dim0_stride[31:0]
    g1[6] = (int)((stride_elems >> 32) & 0xFFFFu);               // stride[47:32]; dim1_stride=0
    g1[7] = 0;
    i32x4 z4 = {0, 0, 0, 0};
#if defined(__clang_major__) && (__clang_major__ >= 23)
    i32x8 z8 = {0, 0, 0, 0, 0, 0, 0, 0};
    __builtin_amdgcn_tensor_load_to_lds(g0, g1, z4, z4, z8, 0);
#else
    __builtin_amdgcn_tensor_load_to_lds(g0, g1, z4, z4, 0);
#endif
}

// ---------------------------------------------------------------- WMMA GEMM, TDM-staged, double-buffered LDS
// Out(RxN) = A(RxK) @ Wt(NxK)^T (+bias). A,Wt f16 row-major; Out f32.
// 256 threads = 8 waves; block tile 128x16; K step 32; K%32==0, N%16==0.
__global__ void k_gemm_wmma(const _Float16* __restrict__ A, const _Float16* __restrict__ Wt,
                            const float* __restrict__ bias, float* __restrict__ Out,
                            int R, int K, int N) {
    __shared__ _Float16 smA[2][128 * TDM_PITCH];
    __shared__ _Float16 smB[2][16 * TDM_PITCH];
    int wave = threadIdx.x >> 5;
    int lane = threadIdx.x & 31;
    int half = lane >> 4;
    int r16  = lane & 15;
    int n0   = blockIdx.x * 16;
    int m0   = blockIdx.y * 128;
    int mrows = R - m0; if (mrows > 128) mrows = 128;   // clamp tile_dim1 (no OOB reliance)
    uint32_t ldsA[2] = { (uint32_t)(uintptr_t)&smA[0][0], (uint32_t)(uintptr_t)&smA[1][0] };
    uint32_t ldsB[2] = { (uint32_t)(uintptr_t)&smB[0][0], (uint32_t)(uintptr_t)&smB[1][0] };
    const _Float16* gA = A  + (size_t)m0 * K;
    const _Float16* gB = Wt + (size_t)n0 * K;

    if (wave == 0) {   // wave-uniform branch; TDM issued once per wave
        tdm_load_2d(ldsA[0], gA, (uint32_t)K, (uint32_t)R, 32, (uint32_t)mrows, (uint64_t)K);
        tdm_load_2d(ldsB[0], gB, (uint32_t)K, (uint32_t)N, 32, 16, (uint64_t)K);
    }

    v8f acc = {};
    int cur = 0;
    for (int kk = 0; kk < K; kk += 32) {
        int nxt = cur ^ 1;
        if (wave == 0) {
            if (kk + 32 < K) {
                tdm_load_2d(ldsA[nxt], gA + kk + 32, (uint32_t)K, (uint32_t)R, 32,
                            (uint32_t)mrows, (uint64_t)K);
                tdm_load_2d(ldsB[nxt], gB + kk + 32, (uint32_t)K, (uint32_t)N, 32, 16,
                            (uint64_t)K);
                __builtin_amdgcn_s_wait_tensorcnt(2);   // allow the 2 just-issued to fly
            } else {
                __builtin_amdgcn_s_wait_tensorcnt(0);
            }
        }
        __syncthreads();                                // cur buffers now valid for all waves
        // A 16x32 f16 fragment: lanes 0-15 hold K {0..7,16..23}; lanes 16-31 K {8..15,24..31}
        const _Float16* la = &smA[cur][(wave * 16 + r16) * TDM_PITCH];
        // B 32x16 f16 fragment: lane = col; same K layout as A
        const _Float16* lb = &smB[cur][r16 * TDM_PITCH];
        v16h a, bf;
        #pragma unroll
        for (int e = 0; e < 8; e++) {
            a[e]     = la[8 * half + e];
            a[e + 8] = la[16 + 8 * half + e];
        }
        #pragma unroll
        for (int e = 0; e < 8; e++) { bf[e] = lb[8 * half + e]; bf[e + 8] = lb[16 + 8 * half + e]; }
        acc = __builtin_amdgcn_wmma_f32_16x16x32_f16(false, a, false, bf, (short)0, acc,
                                                     false, false);
        __syncthreads();                                // done reading before next overwrite
        cur = nxt;
    }

    int mbase = m0 + wave * 16;
    #pragma unroll
    for (int rr = 0; rr < 8; rr++) {
        int m = mbase + rr + 8 * half;   // C/D: VGPR rr -> M=rr (lanes0-15) / M=8+rr (lanes16-31)
        if (m < R) {
            int n = n0 + (lane & 15);
            float v = acc[rr];
            if (bias) v += bias[n];
            Out[(size_t)m * N + n] = v;
        }
    }
}

// ---------------------------------------------------------------- softmax over token axis on k (in place inside qkv)
__global__ void k_softmax_tokens(float* __restrict__ qkv, int N) {
    int c = threadIdx.x;     // 0..255 -> k column 256+c
    int b = blockIdx.x;      // 0..7
    float* base = qkv + (size_t)b * N * 768 + 256 + c;
    float mx = -3.0e38f;
    for (int n = 0; n < N; n++) mx = fmaxf(mx, base[(size_t)n * 768]);
    float s = 0.f;
    for (int n = 0; n < N; n++) s += __expf(base[(size_t)n * 768] - mx);
    float inv = 1.f / s;
    for (int n = 0; n < N; n++) base[(size_t)n * 768] = __expf(base[(size_t)n * 768] - mx) * inv;
}

// ---------------------------------------------------------------- kv[b,h,ck,cv] = sum_n ksm * v
__global__ void k_kv(const float* __restrict__ qkv, float* __restrict__ kv, int N) {
    int t = blockIdx.x * blockDim.x + threadIdx.x;
    if (t >= 8 * 8 * 32 * 32) return;
    int cv = t & 31, ck = (t >> 5) & 31, h = (t >> 10) & 7, b = t >> 13;
    const float* kp = qkv + (size_t)b * N * 768 + 256 + h * 32 + ck;
    const float* vp = qkv + (size_t)b * N * 768 + 512 + h * 32 + cv;
    float acc = 0.f;
    for (int n = 0; n < N; n++) acc += kp[(size_t)n * 768] * vp[(size_t)n * 768];
    kv[t] = acc;
}

// ---------------------------------------------------------------- CRPE depthwise convs on v-map (3x3/5x5/7x7 groups)
__global__ void k_crpe_conv(const float* __restrict__ qkv,
                            const float* __restrict__ w3, const float* __restrict__ b3,
                            const float* __restrict__ w5, const float* __restrict__ b5,
                            const float* __restrict__ w7, const float* __restrict__ b7,
                            float* __restrict__ convv, int N, int H, int W) {
    int c  = threadIdx.x;       // 0..255 (= h*32+cv)
    int bp = blockIdx.x;        // b*HW + p
    int HWl = H * W;
    int b = bp / HWl, p = bp % HWl;
    int y = p / W, x = p % W;
    const float* wv; float bias; int ks;
    if (c < 64)       { ks = 3; wv = w3 + c * 9;          bias = b3[c]; }
    else if (c < 160) { ks = 5; wv = w5 + (c - 64) * 25;  bias = b5[c - 64]; }
    else              { ks = 7; wv = w7 + (c - 160) * 49; bias = b7[c - 160]; }
    int pad = ks >> 1;
    float acc = bias;
    for (int ky = 0; ky < ks; ky++) {
        int iy = y + ky - pad;
        if (iy < 0 || iy >= H) continue;
        for (int kx = 0; kx < ks; kx++) {
            int ix = x + kx - pad;
            if (ix < 0 || ix >= W) continue;
            acc += qkv[((size_t)b * N + 1 + (size_t)iy * W + ix) * 768 + 512 + c] * wv[ky * ks + kx];
        }
    }
    convv[((size_t)b * 256 + c) * HWl + p] = acc;
}

// ---------------------------------------------------------------- att = CH^-.5 * (q @ kv) + q * conv_v (cls: fa only)
__global__ void k_att(const float* __restrict__ qkv, const float* __restrict__ kvm,
                      const float* __restrict__ convv, float* __restrict__ att,
                      int N, int HWl) {
    int cc = threadIdx.x;       // h*32+cv
    int bn = blockIdx.x;        // b*N+n
    int b = bn / N, n = bn % N;
    int h = cc >> 5;
    const float* qrow = qkv + (size_t)bn * 768 + h * 32;
    const float* kvp  = kvm + ((size_t)(b * 8 + h)) * 1024 + (cc & 31);
    float acc = 0.f;
    #pragma unroll
    for (int ck = 0; ck < 32; ck++) acc += qrow[ck] * kvp[ck * 32];
    float out = 0.17677669529663687f * acc;   // 32^-0.5
    if (n > 0) out += qkv[(size_t)bn * 768 + cc] * convv[((size_t)b * 256 + cc) * HWl + (n - 1)];
    att[(size_t)bn * 256 + cc] = out;
}

// ---------------------------------------------------------------- exact GELU, f32 -> f16
__global__ void k_gelu_f16(const float* __restrict__ in, _Float16* __restrict__ out, size_t n) {
    size_t i = (size_t)blockIdx.x * blockDim.x + threadIdx.x;
    if (i < n) {
        float x = in[i];
        out[i] = (_Float16)(0.5f * x * (1.f + erff(x * 0.70710678118654752f)));
    }
}

// ---------------------------------------------------------------- bilinear sample from token-layout tensor
DEV_INLINE float bilin(const float* __restrict__ src, int b, int c, int Hs, int Ws, int Ns,
                       float yy, float xx) {
    int y0 = (int)floorf(yy), x0 = (int)floorf(xx);
    float fy = yy - (float)y0, fx = xx - (float)x0;
    int y1 = y0 + 1, x1 = x0 + 1;
    y0 = min(max(y0, 0), Hs - 1); y1 = min(max(y1, 0), Hs - 1);
    x0 = min(max(x0, 0), Ws - 1); x1 = min(max(x1, 0), Ws - 1);
    const float* p = src + (size_t)b * Ns * 256 + c;
    float v00 = p[(size_t)(1 + y0 * Ws + x0) * 256];
    float v01 = p[(size_t)(1 + y0 * Ws + x1) * 256];
    float v10 = p[(size_t)(1 + y1 * Ws + x0) * 256];
    float v11 = p[(size_t)(1 + y1 * Ws + x1) * 256];
    return v00 * (1.f - fy) * (1.f - fx) + v01 * (1.f - fy) * fx +
           v10 * fy * (1.f - fx) + v11 * fy * fx;
}

// ---------------------------------------------------------------- dst = base + interp(srcA) + interp(srcB)
__global__ void k_mix(const float* __restrict__ base,
                      const float* __restrict__ sA, int HA, int WA, int NA,
                      const float* __restrict__ sB, int HB, int WB, int NB,
                      float* __restrict__ dst, int H, int W, int N) {
    int c  = threadIdx.x;
    int bn = blockIdx.x;
    int b = bn / N, n = bn % N;
    float v = base[(size_t)bn * 256 + c];
    if (n == 0) {
        v += sA[(size_t)b * NA * 256 + c] + sB[(size_t)b * NB * 256 + c];
    } else {
        int p = n - 1, y = p / W, x = p % W;
        float ya = ((float)y + 0.5f) * (float)HA / (float)H - 0.5f;
        float xa = ((float)x + 0.5f) * (float)WA / (float)W - 0.5f;
        float yb = ((float)y + 0.5f) * (float)HB / (float)H - 0.5f;
        float xb = ((float)x + 0.5f) * (float)WB / (float)W - 0.5f;
        v += bilin(sA, b, c, HA, WA, NA, ya, xa) + bilin(sB, b, c, HB, WB, NB, yb, xb);
    }
    dst[(size_t)bn * 256 + c] = v;
}

// ---------------------------------------------------------------- elementwise adds
__global__ void k_add_inplace(float* __restrict__ x, const float* __restrict__ y, size_t n) {
    size_t i = (size_t)blockIdx.x * blockDim.x + threadIdx.x;
    if (i < n) x[i] += y[i];
}
__global__ void k_add_out(const float* __restrict__ a, const float* __restrict__ b,
                          float* __restrict__ o, size_t n) {
    size_t i = (size_t)blockIdx.x * blockDim.x + threadIdx.x;
    if (i < n) o[i] = a[i] + b[i];
}

// =====================================================================
extern "C" void kernel_launch(void* const* d_in, const int* in_sizes, int n_in,
                              void* d_out, int out_size, void* d_ws, size_t ws_size,
                              hipStream_t stream) {
    (void)in_sizes; (void)n_in; (void)out_size; (void)ws_size;
    const int NT[4] = {3137, 785, 197, 50};
    const int HH[4] = {56, 28, 14, 7};
    const int WW[4] = {56, 28, 14, 7};
    const int Bb = 8;
    int R[4], HWp[4];
    for (int i = 0; i < 4; i++) { R[i] = Bb * NT[i]; HWp[i] = HH[i] * WW[i]; }

    const float* x1     = (const float*)d_in[0];
    const float* xin[4] = {(const float*)d_in[1], (const float*)d_in[2],
                           (const float*)d_in[3], (const float*)d_in[4]};
    const float* cpe_w  = (const float*)d_in[5];
    const float* cpe_b  = (const float*)d_in[6];
    const float* n1_g   = (const float*)d_in[7];
    const float* n1_b   = (const float*)d_in[8];
    const float* qkv_w  = (const float*)d_in[9];
    const float* proj_w = (const float*)d_in[10];
    const float* proj_b = (const float*)d_in[11];
    const float* crw3   = (const float*)d_in[12];
    const float* crb3   = (const float*)d_in[13];
    const float* crw5   = (const float*)d_in[14];
    const float* crb5   = (const float*)d_in[15];
    const float* crw7   = (const float*)d_in[16];
    const float* crb7   = (const float*)d_in[17];
    const float* n2_g   = (const float*)d_in[18];
    const float* n2_b   = (const float*)d_in[19];
    const float* fc1_w  = (const float*)d_in[20];
    const float* fc1_b  = (const float*)d_in[21];
    const float* fc2_w  = (const float*)d_in[22];
    const float* fc2_b  = (const float*)d_in[23];

    // ---------------- workspace bump allocator ----------------
    char* base = (char*)d_ws;
    size_t cur_off = 0;
    auto alloc = [&](size_t bytes) -> char* {
        char* p = base + cur_off;
        cur_off += (bytes + 255) & ~(size_t)255;
        return p;
    };
    float* xi[4];  float* curb[4];  float* mixb[4];
    for (int i = 0; i < 4; i++) xi[i]   = (float*)alloc((size_t)R[i] * 256 * 4);
    for (int i = 0; i < 4; i++) curb[i] = (float*)alloc((size_t)R[i] * 256 * 4);
    for (int i = 0; i < 4; i++) mixb[i] = (float*)alloc((size_t)R[i] * 256 * 4);
    _Float16* qkvw16 = (_Float16*)alloc((size_t)4 * 768 * 256 * 2);
    _Float16* projw16 = (_Float16*)alloc((size_t)4 * 256 * 256 * 2);
    _Float16* fc1w16 = (_Float16*)alloc((size_t)1024 * 256 * 2);
    _Float16* fc2w16 = (_Float16*)alloc((size_t)256 * 1024 * 2);
    size_t arena_off = cur_off;
    // attention-phase arena (sized for scale 0, reused for all scales)
    _Float16* lnbf  = (_Float16*)alloc((size_t)R[0] * 256 * 2);
    float*    qkvbuf = (float*)alloc((size_t)R[0] * 768 * 4);
    float*    attbuf = (float*)alloc((size_t)R[0] * 256 * 4);
    float*    convv  = (float*)alloc((size_t)Bb * 256 * HWp[0] * 4);
    _Float16* attbf = (_Float16*)alloc((size_t)R[0] * 256 * 2);
    float*    kvbuf  = (float*)alloc((size_t)65536 * 4);
    // mlp-phase arena (overlaps attention arena; phases are sequential)
    cur_off = arena_off;
    _Float16* ln2bf  = (_Float16*)alloc((size_t)R[0] * 256 * 2);
    float*    hbuf   = (float*)alloc((size_t)R[0] * 1024 * 4);
    _Float16* hbf    = (_Float16*)alloc((size_t)R[0] * 1024 * 2);
    float*    fc2out = (float*)alloc((size_t)R[0] * 256 * 4);

    // ---------------- x1 passthrough ----------------
    size_t x1_elems = (size_t)Bb * (1 + 112 * 112) * 256;
    hipMemcpyAsync(d_out, x1, x1_elems * 4, hipMemcpyDeviceToDevice, stream);

    // ---------------- weight conversion ----------------
    {
        size_t n;
        n = (size_t)4 * 768 * 256;
        k_f32_to_f16<<<(unsigned)((n + 255) / 256), 256, 0, stream>>>(qkv_w, qkvw16, n);
        n = (size_t)4 * 256 * 256;
        k_f32_to_f16<<<(unsigned)((n + 255) / 256), 256, 0, stream>>>(proj_w, projw16, n);
        n = (size_t)1024 * 256;
        k_f32_to_f16<<<(unsigned)((n + 255) / 256), 256, 0, stream>>>(fc1_w, fc1w16, n);
        n = (size_t)256 * 1024;
        k_f32_to_f16<<<(unsigned)((n + 255) / 256), 256, 0, stream>>>(fc2_w, fc2w16, n);
    }

    // ---------------- per-scale attention ----------------
    for (int i = 0; i < 4; i++) {
        int N = NT[i], H = HH[i], W = WW[i], Ri = R[i], HWl = HWp[i];
        size_t rc = (size_t)Ri * 256;
        k_cpe<<<Ri, 256, 0, stream>>>(xin[i], cpe_w + (size_t)i * 256 * 9, cpe_b + i * 256,
                                      xi[i], N, H, W);
        k_ln_f16<<<(Ri + 7) / 8, 256, 0, stream>>>(xi[i], n1_g + i * 256, n1_b + i * 256,
                                                   lnbf, (size_t)Ri);
        dim3 gq(768 / 16, (Ri + 127) / 128);
        k_gemm_wmma<<<gq, 256, 0, stream>>>(lnbf, qkvw16 + (size_t)i * 768 * 256,
                                            nullptr, qkvbuf, Ri, 256, 768);
        k_softmax_tokens<<<Bb, 256, 0, stream>>>(qkvbuf, N);
        k_kv<<<65536 / 256, 256, 0, stream>>>(qkvbuf, kvbuf, N);
        k_crpe_conv<<<Bb * HWl, 256, 0, stream>>>(qkvbuf,
            crw3 + (size_t)i * 64 * 9,  crb3 + i * 64,
            crw5 + (size_t)i * 96 * 25, crb5 + i * 96,
            crw7 + (size_t)i * 96 * 49, crb7 + i * 96,
            convv, N, H, W);
        k_att<<<Ri, 256, 0, stream>>>(qkvbuf, kvbuf, convv, attbuf, N, HWl);
        k_f32_to_f16<<<(unsigned)((rc + 255) / 256), 256, 0, stream>>>(attbuf, attbf, rc);
        dim3 gp(256 / 16, (Ri + 127) / 128);
        k_gemm_wmma<<<gp, 256, 0, stream>>>(attbf, projw16 + (size_t)i * 256 * 256,
                                            proj_b + i * 256, curb[i], Ri, 256, 256);
    }

    // ---------------- cross-scale bilinear mixing ----------------
    k_mix<<<R[0], 256, 0, stream>>>(curb[0], curb[1], HH[1], WW[1], NT[1],
                                    curb[2], HH[2], WW[2], NT[2],
                                    mixb[0], HH[0], WW[0], NT[0]);
    k_mix<<<R[1], 256, 0, stream>>>(curb[1], curb[2], HH[2], WW[2], NT[2],
                                    curb[0], HH[0], WW[0], NT[0],
                                    mixb[1], HH[1], WW[1], NT[1]);
    k_mix<<<R[2], 256, 0, stream>>>(curb[2], curb[3], HH[3], WW[3], NT[3],
                                    curb[0], HH[0], WW[0], NT[0],
                                    mixb[2], HH[2], WW[2], NT[2]);
    k_mix<<<R[3], 256, 0, stream>>>(curb[3], curb[2], HH[2], WW[2], NT[2],
                                    curb[0], HH[0], WW[0], NT[0],
                                    mixb[3], HH[3], WW[3], NT[3]);

    // ---------------- per-scale MLP + residuals ----------------
    size_t out_off = x1_elems;
    for (int i = 0; i < 4; i++) {
        int Ri = R[i];
        size_t rc = (size_t)Ri * 256;
        size_t rh = (size_t)Ri * 1024;
        k_add_inplace<<<(unsigned)((rc + 255) / 256), 256, 0, stream>>>(xi[i], mixb[i], rc);
        k_ln_f16<<<(Ri + 7) / 8, 256, 0, stream>>>(xi[i], n2_g + i * 256, n2_b + i * 256,
                                                   ln2bf, (size_t)Ri);
        dim3 g1(1024 / 16, (Ri + 127) / 128);
        k_gemm_wmma<<<g1, 256, 0, stream>>>(ln2bf, fc1w16, fc1_b, hbuf, Ri, 256, 1024);
        k_gelu_f16<<<(unsigned)((rh + 255) / 256), 256, 0, stream>>>(hbuf, hbf, rh);
        dim3 g2(256 / 16, (Ri + 127) / 128);
        k_gemm_wmma<<<g2, 256, 0, stream>>>(hbf, fc2w16, fc2_b, fc2out, Ri, 1024, 256);
        k_add_out<<<(unsigned)((rc + 255) / 256), 256, 0, stream>>>(
            xi[i], fc2out, (float*)d_out + out_off, rc);
        out_off += rc;
    }
}
